// Foveator_53085795779460
// MI455X (gfx1250) — 8-bit matrix operand order repaired and checked by reading, compiled, not run on hardware
//
#include <hip/hip_runtime.h>


typedef int v8i __attribute__((ext_vector_type(8)));

#define IMG_SHIFT 12
#define CH_SHIFT  24

__device__ __forceinline__ void ring_coord(int j, int& x, int& y) {
  if (j < 64)       { y = j >> 4;                x = j & 15; }
  else if (j < 128) { int t = j - 64; y = 4 + (t >> 3);
                      int w = t & 7;             x = (w < 4) ? w : (w + 8); }
  else              { int t = j - 128; y = 12 + (t >> 4); x = t & 15; }
}

__device__ __forceinline__ unsigned pack4(int x, int y, int z, int w) {
  unsigned xy = __builtin_amdgcn_perm((unsigned)y, (unsigned)x, 0x00000400u);
  unsigned zw = __builtin_amdgcn_perm((unsigned)w, (unsigned)z, 0x00000400u);
  return __builtin_amdgcn_perm(zw, xy, 0x05040100u);
}

__global__ void __launch_bounds__(256)
fovea_level0(const int* __restrict__ img, float* __restrict__ out) {
  int idx = blockIdx.x * 256 + threadIdx.x;
  if (idx >= 256 * 3 * 256) return;
  int n   = idx / 768;
  int rem = idx - n * 768;
  int ch  = rem >> 8;
  int p   = rem & 255;
  int ty  = p >> 4, tx = p & 15;
  int cx  = ((120 + (n & 15)) << 4) + tx;
  int cy  = ((120 + (n >> 4)) << 4) + ty;
  out[idx] = (float)img[((size_t)ch << CH_SHIFT) + ((size_t)cy << IMG_SHIFT) + cx];
}

template <int LEVEL>
__global__ void __launch_bounds__(256)
fovea_pool(const int* __restrict__ img, float* __restrict__ out) {
  constexpr int S       = 1 << LEVEL;
  constexpr int NCHUNKS = (S >= 4) ? (S >> 2) : 1;
  constexpr int NBANDS  = S;
  constexpr int SHIFT   = 2 * LEVEL;
  constexpr int OFF     = (256 - (S << 4)) >> 1;

  const int lane = threadIdx.x & 31;
  const int wave = threadIdx.x >> 5;
  const int gid  = blockIdx.x * 8 + wave;
  const int j    = gid / 3;
  const int ch   = gid - j * 3;
  const int n    = 256 + (LEVEL - 1) * 192 + j;

  int gx, gy; ring_coord(j, gx, gy);
  const int cx = (OFF + S * gx) << 4;
  const int cy = (OFF + S * gy) << 4;

  const int half = lane >> 4;
  const int m    = lane & 15;

  const int* chImg  = img + ((size_t)ch << CH_SHIFT);
  float*     outTok = out + ((size_t)n * 3 + ch) * 256;

  v8i Bf[NCHUNKS];
#pragma unroll
  for (int chunk = 0; chunk < NCHUNKS; ++chunk) {
#pragma unroll
    for (int v = 0; v < 8; ++v) {
      const int kg = (chunk << 6) + ((v >> 1) << 4) + (half << 3) + ((v & 1) << 2);
      unsigned bb;
      if (LEVEL == 1) {
        unsigned lo = (kg     < 32 && ((kg    ) >> 1) == m) ? 0x00000101u : 0u;
        unsigned hi = (kg + 2 < 32 && ((kg + 2) >> 1) == m) ? 0x01010000u : 0u;
        bb = lo | hi;
      } else {
        bb = ((kg >> LEVEL) == m) ? 0x01010101u : 0u;
      }
      Bf[chunk][v] = (int)bb;
    }
  }

  for (int band = 0; band < NBANDS; ++band) {
    v8i acc = {0, 0, 0, 0, 0, 0, 0, 0};
    const int  rowY   = cy + (band << 4) + m;
    const int* rowPtr = chImg + ((size_t)rowY << IMG_SHIFT) + cx;

    if (band + 1 < NBANDS)
      __builtin_prefetch(rowPtr + (16 << IMG_SHIFT), 0, 1);

#pragma unroll
    for (int chunk = 0; chunk < NCHUNKS; ++chunk) {
      v8i a;
#pragma unroll
      for (int v = 0; v < 8; ++v) {
        const int kb = ((v >> 1) << 4) + (half << 3) + ((v & 1) << 2);
        const int4 p = *(const int4*)(rowPtr + (chunk << 6) + kb);
        a[v] = (int)pack4(p.x, p.y, p.z, p.w);
      }
      acc = __builtin_amdgcn_wmma_i32_16x16x64_iu8(false, a, false, Bf[chunk],
                                                   acc, false, false);
    }

    if (LEVEL == 1) {
      int t0 = acc[0] + acc[1], t1 = acc[2] + acc[3];
      int t2 = acc[4] + acc[5], t3 = acc[6] + acc[7];
      int rbase = (band << 3) + (half << 2);
      outTok[(rbase + 0) * 16 + m] = (float)(t0 >> SHIFT);
      outTok[(rbase + 1) * 16 + m] = (float)(t1 >> SHIFT);
      outTok[(rbase + 2) * 16 + m] = (float)(t2 >> SHIFT);
      outTok[(rbase + 3) * 16 + m] = (float)(t3 >> SHIFT);
    } else if (LEVEL == 2) {
      int s0 = acc[0] + acc[1] + acc[2] + acc[3];
      int s1 = acc[4] + acc[5] + acc[6] + acc[7];
      int rbase = (band << 2) + (half << 1);
      outTok[(rbase + 0) * 16 + m] = (float)(s0 >> SHIFT);
      outTok[(rbase + 1) * 16 + m] = (float)(s1 >> SHIFT);
    } else if (LEVEL == 3) {
      int ss = acc[0] + acc[1] + acc[2] + acc[3] + acc[4] + acc[5] + acc[6] + acc[7];
      int r = (band << 1) + half;
      outTok[r * 16 + m] = (float)(ss >> SHIFT);
    } else {
      int ss = acc[0] + acc[1] + acc[2] + acc[3] + acc[4] + acc[5] + acc[6] + acc[7];
      ss += __builtin_amdgcn_ds_swizzle(ss, 0x401F);
      if (half == 0) outTok[band * 16 + m] = (float)(ss >> SHIFT);
    }
  }
}

extern "C" void kernel_launch(void* const* d_in, const int* in_sizes, int n_in,
                              void* d_out, int out_size, void* d_ws, size_t ws_size,
                              hipStream_t stream) {
  const int* img = (const int*)d_in[0];
  float*     out = (float*)d_out;

  fovea_level0<<<768, 256, 0, stream>>>(img, out);
  fovea_pool<1><<<72, 256, 0, stream>>>(img, out);
  fovea_pool<2><<<72, 256, 0, stream>>>(img, out);
  fovea_pool<3><<<72, 256, 0, stream>>>(img, out);
  fovea_pool<4><<<72, 256, 0, stream>>>(img, out);
}
